// CrossAttention_72490458021902
// MI455X (gfx1250) — matrix operand order repaired and checked by reading, not compiled, not run on hardware
//
#include <hip/hip_runtime.h>
#include <hip/hip_bf16.h>

// ---------------------------------------------------------------------------
// CDNA5 (gfx1250) cross-attention, bf16 WMMA (16x16x32) with f32 accumulation.
// LDS-staged; A and B operands both read as 2x ds_load_b128 per WMMA.
// ---------------------------------------------------------------------------

typedef __attribute__((ext_vector_type(16))) __bf16        v16bf;
typedef __attribute__((ext_vector_type(8)))  float         v8f;
typedef __attribute__((ext_vector_type(4)))  float         f32x4;
typedef __attribute__((ext_vector_type(4)))  unsigned int  u32x4;
typedef unsigned int uint_t;

union BFV { v16bf v; unsigned short u[16]; unsigned p[8]; u32x4 q[2]; };

// float -> bf16 bits, round-to-nearest-even (fallback path)
static __device__ inline unsigned short f2bf_u(float f) {
    union { float f; unsigned u; } x; x.f = f;
    unsigned r = x.u + 0x7FFFu + ((x.u >> 16) & 1u);
    return (unsigned short)(r >> 16);
}
// pack two floats into one u32 holding two bf16 (v_cvt_pk_bf16_f32 if present)
static __device__ inline unsigned pk2bf(float lo, float hi) {
#if __has_builtin(__builtin_amdgcn_cvt_pk_bf16_f32)
    auto r = __builtin_amdgcn_cvt_pk_bf16_f32(lo, hi);
    unsigned out; __builtin_memcpy(&out, &r, 4);
    return out;
#else
    return (unsigned)f2bf_u(lo) | ((unsigned)f2bf_u(hi) << 16);
#endif
}

// A-operand (16x32 bf16): lane l holds row (l&15);
// element i -> k = (l>=16?16:0) + i (contiguous, matches B); pairs are consecutive k.
static __device__ inline int aKp(int j, int l) {        // k of element 2j
    return ((l >> 4) << 4) + (j << 1);
}
// B-operand: lane l holds col (l&15); elements are 16 consecutive k starting
// at (l>=16?16:0). With Bt[n][k] (column-major) this is 2 contiguous b128.
static __device__ inline void loadB(BFV& bb, const unsigned short* base) {
    bb.q[0] = *(const u32x4*)base;
    bb.q[1] = *(const u32x4*)(base + 8);
}
static __device__ inline void loadA(BFV& a, const unsigned short* rowbase, int l) {
    #pragma unroll
    for (int j = 0; j < 8; ++j)
        a.p[j] = *(const uint_t*)(rowbase + aKp(j, l));
}

static __device__ inline v8f wmma_bf16(const BFV& a, const BFV& b, v8f c) {
    return __builtin_amdgcn_wmma_f32_16x16x32_bf16(
        false, a.v, false, b.v, (short)0, c, false, false);
}

// Problem constants
#define BATCH 16
#define NQ    4096
#define MKV   77
#define HEADS 8
#define DH    40
#define QD    320
#define CD    768
#define IDIM  320
#define MP    96      // padded KV length (3 k-steps of 32)
#define DP    64      // padded head dim (2 k-steps of 32)

// ---------------------------------------------------------------------------
// Kernel 1: K/V projection. context[b] (77x768) @ W (768x320), written as:
//   kv==0 : Kp[b][h][j(96)][d(64)]  (natural, zero padded)   -- B for scores
//   kv==1 : Vt[b][h][d(64)][j(96)]  (transposed, zero padded) -- B for attn@V
// ---------------------------------------------------------------------------
__global__ __launch_bounds__(256) void kv_proj_kernel(
    const float* __restrict__ ctx, const float* __restrict__ Wk,
    const float* __restrict__ Wv, unsigned short* __restrict__ Kp,
    unsigned short* __restrict__ Vt)
{
    int b  = blockIdx.x >> 1;
    int kv = blockIdx.x & 1;
    int tid = threadIdx.x, l = tid & 31, w = tid >> 5;
    const float* W = kv ? Wv : Wk;

    __shared__ __align__(16) unsigned short ctxS[80 * 32];   // A chunk, row-major
    __shared__ __align__(16) unsigned short wST[320 * 32];   // B chunk, col-major

    // zero-fill this batch's padded slice
    {
        u32x4 z = {0u, 0u, 0u, 0u};
        u32x4* base = (u32x4*)((kv == 0) ? (Kp + (size_t)b * HEADS * MP * DP)
                                         : (Vt + (size_t)b * HEADS * DP * MP));
        for (int i = tid; i < (HEADS * DP * MP) / 8; i += 256) base[i] = z;
    }

    v8f c[13];
    #pragma unroll
    for (int j = 0; j < 13; ++j) c[j] = v8f{};

    for (int kk = 0; kk < 24; ++kk) {
        int k0 = kk * 32;
        __syncthreads();
        // stage context chunk [80][32] row-major (rows >= 77 zero)
        for (int i = tid; i < 640; i += 256) {
            int r = i >> 3, kq = i & 7;
            f32x4 v = {0.f, 0.f, 0.f, 0.f};
            if (r < MKV)
                v = *(const f32x4*)(ctx + (size_t)b * MKV * CD + (size_t)r * CD + k0 + kq * 4);
            ((uint_t*)ctxS)[r * 16 + kq * 2]     = pk2bf(v[0], v[1]);
            ((uint_t*)ctxS)[r * 16 + kq * 2 + 1] = pk2bf(v[2], v[3]);
        }
        // stage W chunk col-major wST[n(320)][k(32)]: two k-rows per iteration
        for (int i = tid; i < 1280; i += 256) {
            int kp = i / 80, nq = i % 80;
            const float* p = W + (size_t)(k0 + kp * 2) * IDIM + nq * 4;
            f32x4 v0 = *(const f32x4*)p;
            f32x4 v1 = *(const f32x4*)(p + IDIM);
            #pragma unroll
            for (int j = 0; j < 4; ++j)
                ((uint_t*)wST)[(nq * 4 + j) * 16 + kp] = pk2bf(v0[j], v1[j]);
        }
        __syncthreads();
        #pragma unroll
        for (int j = 0; j < 13; ++j) {
            int t = w + 8 * j;
            if (t < 100) {
                int tr = t / 20, tc = t % 20;
                BFV a, bb;
                loadA(a, &ctxS[(tr * 16 + (l & 15)) * 32], l);
                loadB(bb, &wST[(tc * 16 + (l & 15)) * 32 + ((l >> 4) << 4)]);
                c[j] = wmma_bf16(a, bb, c[j]);
            }
        }
    }
    __syncthreads();
    #pragma unroll
    for (int j = 0; j < 13; ++j) {
        int t = w + 8 * j;
        if (t < 100) {
            int tr = t / 20, tc = t % 20;
            int n = tc * 16 + (l & 15);
            int h = n / DH, d = n % DH;
            #pragma unroll
            for (int g = 0; g < 8; ++g) {
                int jrow = tr * 16 + g + ((l >> 4) << 3);
                float val = (jrow < MKV) ? c[j][g] : 0.f;
                if (kv == 0)
                    Kp[(((size_t)b * HEADS + h) * MP + jrow) * DP + d] = f2bf_u(val);
                else
                    Vt[(((size_t)b * HEADS + h) * DP + d) * MP + jrow] = f2bf_u(val);
            }
        }
    }
}

// ---------------------------------------------------------------------------
// Kernel 2: fused per-(b, h, 64-row block) attention.
// LDS arena (56KB), phase-overlaid:
//   [0,24K)    : phase1 xs[64][32], wshT[64][32]  -> phase2/3 Ss f32[64][96]
//   [24K,32K)  : Qs bf16[64][64]
//   [32K,44K)  : phase2 KpS bf16[96][64] -> phase4 VtS bf16[64][96]
//   [44K,56K)  : As bf16[64][96]
// ---------------------------------------------------------------------------
__global__ __launch_bounds__(128) void attn_kernel(
    const float* __restrict__ x, const float* __restrict__ Wq,
    const unsigned short* __restrict__ Kp, const unsigned short* __restrict__ Vt,
    unsigned short* __restrict__ Ow)
{
    int idx = blockIdx.x;
    int b  = idx >> 9;          // / (HEADS * 64)
    int h  = (idx >> 6) & 7;
    int rb = idx & 63;          // 64-row block within N
    int tid = threadIdx.x, l = tid & 31, w = tid >> 5;

    __shared__ __align__(16) unsigned char smem[57344];
    unsigned short* xs   = (unsigned short*)smem;              // [64][32] A chunk
    unsigned short* wshT = (unsigned short*)(smem + 4096);     // [64][32] B col-major
    float*          Ss   = (float*)smem;                       // [64][96]
    unsigned short* Qs   = (unsigned short*)(smem + 24576);    // [64][64]
    unsigned short* KVs  = (unsigned short*)(smem + 32768);    // KpS / VtS (12KB)
    unsigned short* As   = (unsigned short*)(smem + 45056);    // [64][96]

    const float* xrow = x + ((size_t)b * NQ + rb * 64) * QD;
    int kb16 = (l >> 4) << 4;   // B-operand k base for this lane

    // ---- Phase 1: Q_h = x_blk @ Wq[:, h*40:+40]  (K=320, 10 chunks) ----
    v8f q[4] = {v8f{}, v8f{}, v8f{}, v8f{}};
    for (int kc = 0; kc < 10; ++kc) {
        int k0 = kc * 32;
        // stage x chunk [64][32] row-major, packed bf16
        for (int i = tid; i < 512; i += 128) {
            int r = i >> 3, kq = i & 7;
            f32x4 v = *(const f32x4*)(xrow + (size_t)r * QD + k0 + kq * 4);
            ((uint_t*)xs)[r * 16 + kq * 2]     = pk2bf(v[0], v[1]);
            ((uint_t*)xs)[r * 16 + kq * 2 + 1] = pk2bf(v[2], v[3]);
            if (kc < 9) __builtin_prefetch(xrow + (size_t)r * QD + k0 + 32, 0, 1);
        }
        // stage Wq_h chunk col-major wshT[n(64)][k(32)], cols >= 40 zero
        for (int i = tid; i < 160; i += 128) {
            int kp = i / 10, nq = i % 10;
            const float* p = Wq + (size_t)(k0 + kp * 2) * IDIM + h * DH + nq * 4;
            f32x4 v0 = *(const f32x4*)p;
            f32x4 v1 = *(const f32x4*)(p + IDIM);
            #pragma unroll
            for (int j = 0; j < 4; ++j)
                ((uint_t*)wshT)[(nq * 4 + j) * 16 + kp] = pk2bf(v0[j], v1[j]);
        }
        for (int i = tid; i < 384; i += 128)        // rows 40..63 zero
            ((uint_t*)wshT)[40 * 16 + i] = 0u;
        __syncthreads();
        BFV a, bb;
        loadA(a, &xs[(w * 16 + (l & 15)) * 32], l);
        #pragma unroll
        for (int tc = 0; tc < 4; ++tc) {
            loadB(bb, &wshT[(tc * 16 + (l & 15)) * 32 + kb16]);
            q[tc] = wmma_bf16(a, bb, q[tc]);
        }
        __syncthreads();
    }
    // write Qs (bf16) and stage K head-slice (12KB) cooperatively
    #pragma unroll
    for (int tc = 0; tc < 4; ++tc)
        #pragma unroll
        for (int g = 0; g < 8; ++g)
            Qs[(w * 16 + g + ((l >> 4) << 3)) * 64 + tc * 16 + (l & 15)] = f2bf_u(q[tc][g]);
    {
        const u32x4* src = (const u32x4*)(Kp + ((size_t)(b * HEADS + h)) * MP * DP);
        for (int i = tid; i < (MP * DP) / 8; i += 128) ((u32x4*)KVs)[i] = src[i];
    }
    __syncthreads();

    // ---- Phase 2: S = Qs(64x64) @ K^T via KpS[96][64] col-major B ----
    {
        v8f s[6] = {v8f{}, v8f{}, v8f{}, v8f{}, v8f{}, v8f{}};
        for (int ks = 0; ks < 2; ++ks) {
            int k0 = ks * 32;
            BFV a, bb;
            loadA(a, &Qs[(w * 16 + (l & 15)) * 64 + k0], l);
            #pragma unroll
            for (int tc = 0; tc < 6; ++tc) {
                loadB(bb, &KVs[(tc * 16 + (l & 15)) * 64 + k0 + kb16]);
                s[tc] = wmma_bf16(a, bb, s[tc]);
            }
        }
        const float scale = 0.15811388300841898f; // 40^-0.5
        #pragma unroll
        for (int tc = 0; tc < 6; ++tc)
            #pragma unroll
            for (int g = 0; g < 8; ++g)
                Ss[(w * 16 + g + ((l >> 4) << 3)) * 96 + tc * 16 + (l & 15)] = s[tc][g] * scale;
    }
    __syncthreads();

    // ---- Phase 3: softmax over j<77 (one row/thread) + stage V^T slice ----
    if (tid < 64) {
        float mx = -1e30f;
        for (int j = 0; j < MKV; ++j) mx = fmaxf(mx, Ss[tid * 96 + j]);
        float sum = 0.f;
        for (int j = 0; j < MKV; ++j) sum += __expf(Ss[tid * 96 + j] - mx);
        float inv = 1.f / sum;
        for (int jq = 0; jq < 48; ++jq) {
            int j0 = 2 * jq, j1 = 2 * jq + 1;
            float e0 = (j0 < MKV) ? __expf(Ss[tid * 96 + j0] - mx) * inv : 0.f;
            float e1 = (j1 < MKV) ? __expf(Ss[tid * 96 + j1] - mx) * inv : 0.f;
            ((uint_t*)As)[tid * 48 + jq] = pk2bf(e0, e1);
        }
    }
    {
        const u32x4* src = (const u32x4*)(Vt + ((size_t)(b * HEADS + h)) * DP * MP);
        for (int i = tid; i < (MP * DP) / 8; i += 128) ((u32x4*)KVs)[i] = src[i];
    }
    __syncthreads();

    // ---- Phase 4: O = As(64x96) @ V via VtS[64][96] col-major B ----
    {
        v8f o[4] = {v8f{}, v8f{}, v8f{}, v8f{}};
        for (int ks = 0; ks < 3; ++ks) {
            int k0 = ks * 32;
            BFV a, bb;
            loadA(a, &As[(w * 16 + (l & 15)) * 96 + k0], l);
            #pragma unroll
            for (int tc = 0; tc < 4; ++tc) {
                loadB(bb, &KVs[(tc * 16 + (l & 15)) * 96 + k0 + kb16]);
                o[tc] = wmma_bf16(a, bb, o[tc]);
            }
        }
        #pragma unroll
        for (int tc = 0; tc < 4; ++tc) {
            int d = tc * 16 + (l & 15);
            if (d < DH) {
                #pragma unroll
                for (int g = 0; g < 8; ++g) {
                    int row = rb * 64 + w * 16 + g + ((l >> 4) << 3);
                    Ow[((size_t)b * NQ + row) * IDIM + h * DH + d] = f2bf_u(o[tc][g]);
                }
            }
        }
    }
}

// ---------------------------------------------------------------------------
// Kernel 3: out = Ow(bf16) @ Wo + bo  (f32 out). 64 rows/block, 8 waves.
// ---------------------------------------------------------------------------
__global__ __launch_bounds__(256) void oproj_kernel(
    const unsigned short* __restrict__ Ow, const float* __restrict__ Wo,
    const float* __restrict__ bo, float* __restrict__ out)
{
    int gr = blockIdx.x;   // 1024 blocks of 64 rows
    int tid = threadIdx.x, l = tid & 31, w = tid >> 5;
    int tr = w >> 1, half = w & 1;
    int kb16 = (l >> 4) << 4;

    __shared__ __align__(16) unsigned short os[64 * 32];    // A chunk (bf16)
    __shared__ __align__(16) unsigned short wcT[320 * 32];  // B col-major

    v8f c[10];
    #pragma unroll
    for (int j = 0; j < 10; ++j) c[j] = v8f{};

    for (int kc = 0; kc < 10; ++kc) {
        int k0 = kc * 32;
        // stage O chunk [64][32] bf16 (straight b128 copy)
        {
            int r = tid >> 2, qz = tid & 3;   // 256 x u32x4
            ((u32x4*)os)[tid] =
                *(const u32x4*)(Ow + ((size_t)gr * 64 + r) * IDIM + k0 + qz * 8);
        }
        // stage Wo chunk col-major wcT[n(320)][k(32)]
        for (int i = tid; i < 1280; i += 256) {
            int kp = i / 80, nq = i % 80;
            const float* p = Wo + (size_t)(k0 + kp * 2) * QD + nq * 4;
            f32x4 v0 = *(const f32x4*)p;
            f32x4 v1 = *(const f32x4*)(p + QD);
            #pragma unroll
            for (int j = 0; j < 4; ++j)
                ((uint_t*)wcT)[(nq * 4 + j) * 16 + kp] = pk2bf(v0[j], v1[j]);
            if (kc < 9) __builtin_prefetch(p + 32 * QD, 0, 1);
        }
        __syncthreads();
        BFV a;
        loadA(a, &os[(tr * 16 + (l & 15)) * 32], l);
        #pragma unroll
        for (int j = 0; j < 10; ++j) {
            BFV bb;
            int tc = half * 10 + j;
            loadB(bb, &wcT[(tc * 16 + (l & 15)) * 32 + kb16]);
            c[j] = wmma_bf16(a, bb, c[j]);
        }
        __syncthreads();
    }
    #pragma unroll
    for (int j = 0; j < 10; ++j) {
        int tc = half * 10 + j;
        int col = tc * 16 + (l & 15);
        float bias = bo[col];
        #pragma unroll
        for (int g = 0; g < 8; ++g) {
            int row = gr * 64 + tr * 16 + g + ((l >> 4) << 3);
            out[(size_t)row * QD + col] = c[j][g] + bias;
        }
    }
}

// ---------------------------------------------------------------------------
extern "C" void kernel_launch(void* const* d_in, const int* in_sizes, int n_in,
                              void* d_out, int out_size, void* d_ws, size_t ws_size,
                              hipStream_t stream) {
    const float* x   = (const float*)d_in[0];
    const float* ctx = (const float*)d_in[1];
    const float* Wq  = (const float*)d_in[2];
    const float* Wk  = (const float*)d_in[3];
    const float* Wv  = (const float*)d_in[4];
    const float* Wo  = (const float*)d_in[5];
    const float* bo  = (const float*)d_in[6];
    float* out = (float*)d_out;

    // workspace: Kp (1.5MB) | Vt (1.5MB) | Ow (40MB), all bf16 bits
    unsigned short* Kp = (unsigned short*)d_ws;
    unsigned short* Vt = Kp + (size_t)BATCH * HEADS * MP * DP;
    unsigned short* Ow = Vt + (size_t)BATCH * HEADS * DP * MP;

    kv_proj_kernel<<<dim3(BATCH * 2), dim3(256), 0, stream>>>(ctx, Wk, Wv, Kp, Vt);
    attn_kernel<<<dim3(BATCH * HEADS * (NQ / 64)), dim3(128), 0, stream>>>(x, Wq, Kp, Vt, Ow);
    oproj_kernel<<<dim3((BATCH * NQ) / 64), dim3(256), 0, stream>>>(Ow, Wo, bo, out);
}
